// AEDecoder_10926396801073
// MI455X (gfx1250) — matrix operand order repaired and checked by reading, not compiled, not run on hardware
//
#include <hip/hip_runtime.h>
#include <stdint.h>

// Problem constants (from the reference)
#define B_ROWS   512
#define IN_F     1639
#define OUT_F    17000
#define E_EDGES  400000
#define KP       1664      // IN_F padded to multiple of 32 (WMMA K)
#define NP       17024     // OUT_F padded to multiple of 128 (block N tile)
#define NEG_SLOPE 0.01f

#define LDS_STRIDE 40      // bf16 elements per LDS row (80B: 16B-aligned, bank-spread)
#define NITER (KP / 32)    // 52 K-steps

typedef __bf16 v16bf __attribute__((ext_vector_type(16)));
typedef float  v8f   __attribute__((ext_vector_type(8)));

union Frag16 {
    v16bf v;
    uint4 u[2];
};

__device__ __forceinline__ unsigned short f32_to_bf16_rne(float f) {
    union { float f; uint32_t u; } x;
    x.f = f;
    uint32_t u = x.u;
    return (unsigned short)((u + 0x7FFFu + ((u >> 16) & 1u)) >> 16);
}

// ---------------------------------------------------------------------------
// 1) Zero the padded dense-bf16 weight matrix (56.7 MB -> L2 resident)
// ---------------------------------------------------------------------------
__global__ void zero_w_kernel(uint4* __restrict__ p, long n16) {
    long i = (long)blockIdx.x * blockDim.x + threadIdx.x;
    if (i < n16) {
        uint4 z = {0u, 0u, 0u, 0u};
        p[i] = z;
    }
}

// ---------------------------------------------------------------------------
// 2) Convert features fp32 [512][1639] -> bf16 padded [512][KP]
// ---------------------------------------------------------------------------
__global__ void convert_x_kernel(const float* __restrict__ x,
                                 unsigned short* __restrict__ xb) {
    int i = blockIdx.x * blockDim.x + threadIdx.x;   // over B_ROWS*KP
    if (i >= B_ROWS * KP) return;
    int b = i / KP;
    int k = i - b * KP;
    float v = (k < IN_F) ? x[(size_t)b * IN_F + k] : 0.0f;
    xb[i] = f32_to_bf16_rne(v);
}

// ---------------------------------------------------------------------------
// 3) Scatter edges into dense W[o][k] (bf16) with packed-bf16 atomic adds.
//    Duplicate (o,k) edges must SUM (segment_sum semantics) -> atomic.
//    global_atomic_pk_add_bf16 adds a {lo,hi} bf16 pair; the untouched half
//    gets +0.0 which is a no-op.
// ---------------------------------------------------------------------------
__global__ void scatter_w_kernel(const float* __restrict__ w,
                                 const int* __restrict__ eo,
                                 const int* __restrict__ ei,
                                 unsigned short* __restrict__ wb) {
    int e = blockIdx.x * blockDim.x + threadIdx.x;
    if (e >= E_EDGES) return;
    int o = eo[e];
    int k = ei[e];
    size_t idx = (size_t)o * KP + (size_t)k;
    uint32_t bits = (uint32_t)f32_to_bf16_rne(w[e]);
    uint32_t data = (idx & 1) ? (bits << 16) : bits;        // place in correct half
    uint64_t addr = (uint64_t)(uintptr_t)(wb + (idx & ~(size_t)1)); // 4B aligned
    asm volatile("global_atomic_pk_add_bf16 %0, %1, off"
                 :
                 : "v"(addr), "v"(data)
                 : "memory");
}

// ---------------------------------------------------------------------------
// 4) Tiled WMMA GEMM with ASYNC global->LDS double buffering:
//      out[b,o] = X[b,:] . W[o,:] + bias, LeakyReLU
//    Block tile 128(M) x 128(N), K step 32.  8 waves = 4(M) x 2(N),
//    each wave: 2 m-frags x 4 n-frags of v_wmma_f32_16x16x32_bf16.
//    Tile staging uses GLOBAL_LOAD_ASYNC_TO_LDS_B128 (ASYNCcnt) so no VGPR
//    staging and the wave keeps issuing WMMAs while the next tile streams in.
// ---------------------------------------------------------------------------
__global__ __launch_bounds__(256) void gemm_bias_lrelu_kernel(
        const unsigned short* __restrict__ Xb,   // [B_ROWS][KP] bf16
        const unsigned short* __restrict__ Wb,   // [NP][KP] bf16
        const float* __restrict__ bias,          // [OUT_F]
        float* __restrict__ out) {               // [B_ROWS][OUT_F]
    // lds[buf][0]=A tile, lds[buf][1]=B tile; 2 x 2 x 10240B = 40 KB
    __shared__ unsigned short lds[2][2][128 * LDS_STRIDE];

    const int tid  = threadIdx.x;
    const int lane = tid & 31;
    const int wv   = tid >> 5;        // 0..7
    const int wm   = wv >> 1;         // 0..3  -> M offset wm*32
    const int wn   = wv & 1;          // 0..1  -> N offset wn*64

    const int mbase = blockIdx.y * 128;
    const int nbase = blockIdx.x * 128;

    v8f acc[2][4];
#pragma unroll
    for (int i = 0; i < 2; i++)
#pragma unroll
        for (int j = 0; j < 4; j++) {
            v8f z = {0.f, 0.f, 0.f, 0.f, 0.f, 0.f, 0.f, 0.f};
            acc[i][j] = z;
        }

    // cooperative tile staging: 256 threads, each moves 32B of A and 32B of B
    const int lrow  = tid >> 1;          // 0..127 tile row
    const int lhalf = (tid & 1) * 16;    // element offset 0 or 16

    const int lhi = lane >> 4;           // 0/1 half-wave
    const int lm  = lane & 15;
    const int kbA = lhi * 8;             // A frag K base within tile
    const int sB  = lhi * 8;             // B frag K base within tile

    // Per-thread stream bases (advance 64B per K-step)
    const uint64_t gA0 = (uint64_t)(uintptr_t)(Xb + (size_t)(mbase + lrow) * KP + lhalf);
    const uint64_t gB0 = (uint64_t)(uintptr_t)(Wb + (size_t)(nbase + lrow) * KP + lhalf);
    // LDS byte offsets: flat shared pointer keeps the LDS offset in addr[31:0]
    const uint32_t tileBytes = 128 * LDS_STRIDE * 2;   // 10240
    const uint32_t stThread  = (uint32_t)((lrow * LDS_STRIDE + lhalf) * 2);
    const uint32_t ldsBase   = (uint32_t)(uintptr_t)(&lds[0][0][0]);

    // Issue one stage: 4 async b128 per thread (A lo/hi, B lo/hi).
    // INST_OFFSET applies to BOTH the LDS and global addresses, so the +16B
    // chunk reuses the same address VGPRs with offset:16.
    auto issue_stage = [&](int stage) {
        const uint32_t buf = (uint32_t)(stage & 1);
        const uint64_t ga  = gA0 + (uint64_t)stage * 64;
        const uint64_t gb  = gB0 + (uint64_t)stage * 64;
        const uint32_t la  = ldsBase + buf * 2u * tileBytes + stThread;
        const uint32_t lb  = la + tileBytes;
        asm volatile(
            "global_load_async_to_lds_b128 %0, %2, off\n\t"
            "global_load_async_to_lds_b128 %0, %2, off offset:16\n\t"
            "global_load_async_to_lds_b128 %1, %3, off\n\t"
            "global_load_async_to_lds_b128 %1, %3, off offset:16"
            :
            : "v"(la), "v"(lb), "v"(ga), "v"(gb)
            : "memory");
    };

    issue_stage(0);   // prologue: stage 0 in flight

    for (int it = 0; it < NITER; ++it) {
        const int cur = it & 1;
        if (it + 1 < NITER) {
            issue_stage(it + 1);
            // async loads retire in order: <=4 outstanding means stage `it` done
            asm volatile("s_wait_asynccnt 0x4" ::: "memory");
        } else {
            asm volatile("s_wait_asynccnt 0x0" ::: "memory");
        }
        __syncthreads();   // current tile visible to all waves

        Frag16 afr[2], bfr[4];
#pragma unroll
        for (int mi = 0; mi < 2; mi++) {
            // A 16x32 bf16: lane -> row M=lm, elems [0..7]=K[kbA..], [8..15]=K[kbA+16..]
            const unsigned short* pa =
                &lds[cur][0][(wm * 32 + mi * 16 + lm) * LDS_STRIDE + kbA];
            afr[mi].u[0] = *(const uint4*)(pa);        // K kbA..kbA+7
            afr[mi].u[1] = *(const uint4*)(pa + 16);   // K kbA+16..kbA+23
        }
#pragma unroll
        for (int ni = 0; ni < 4; ni++) {
            // B 32x16 bf16: lane -> col N=lm, elems [0..7]=K[sB..], [8..15]=K[sB+16..]
            const unsigned short* pb =
                &lds[cur][1][(wn * 64 + ni * 16 + lm) * LDS_STRIDE + sB];
            bfr[ni].u[0] = *(const uint4*)(pb);
            bfr[ni].u[1] = *(const uint4*)(pb + 16);
        }
#pragma unroll
        for (int mi = 0; mi < 2; mi++)
#pragma unroll
            for (int ni = 0; ni < 4; ni++)
                acc[mi][ni] = __builtin_amdgcn_wmma_f32_16x16x32_bf16(
                    /*neg_a=*/false, afr[mi].v,
                    /*neg_b=*/false, bfr[ni].v,
                    /*c_mod=*/(short)0, acc[mi][ni],
                    /*reuse_a=*/false, /*reuse_b=*/false);

        __syncthreads();   // all waves done reading buf `cur` before it is refilled
    }

    // Epilogue: C/D layout -> lane l holds N=l&15, M = r + 8*(l>=16) for VGPR r
#pragma unroll
    for (int mi = 0; mi < 2; mi++) {
#pragma unroll
        for (int ni = 0; ni < 4; ni++) {
            int o = nbase + wn * 64 + ni * 16 + lm;
            if (o < OUT_F) {
                float bv = bias[o];
                int mrow = mbase + wm * 32 + mi * 16 + lhi * 8;
#pragma unroll
                for (int r = 0; r < 8; r++) {
                    float v = acc[mi][ni][r] + bv;
                    v = (v >= 0.f) ? v : NEG_SLOPE * v;
                    out[(size_t)(mrow + r) * OUT_F + o] = v;
                }
            }
        }
    }
}

// ---------------------------------------------------------------------------
// Host launcher
// ---------------------------------------------------------------------------
extern "C" void kernel_launch(void* const* d_in, const int* in_sizes, int n_in,
                              void* d_out, int out_size, void* d_ws, size_t ws_size,
                              hipStream_t stream) {
    const float* features = (const float*)d_in[0];   // [512][1639] fp32
    const float* weights  = (const float*)d_in[1];   // [400000] fp32
    const float* bias     = (const float*)d_in[2];   // [17000] fp32
    const int*   edge_out = (const int*)d_in[3];     // [400000]
    const int*   edge_in  = (const int*)d_in[4];     // [400000]
    float*       out      = (float*)d_out;           // [512][17000] fp32

    // workspace layout (bf16 stored as ushort):
    //   Xb : [512][KP]   = 1.70 MB
    //   Wb : [NP][KP]    = 56.7 MB   (fits in 192 MB L2)
    unsigned short* Xb = (unsigned short*)d_ws;
    unsigned short* Wb = Xb + (size_t)B_ROWS * KP;

    const size_t wElems = (size_t)NP * KP;
    const long   n16    = (long)(wElems * sizeof(unsigned short) / 16);

    zero_w_kernel<<<(int)((n16 + 255) / 256), 256, 0, stream>>>((uint4*)Wb, n16);
    convert_x_kernel<<<(B_ROWS * KP + 255) / 256, 256, 0, stream>>>(features, Xb);
    scatter_w_kernel<<<(E_EDGES + 255) / 256, 256, 0, stream>>>(weights, edge_out, edge_in, Wb);

    dim3 grid(NP / 128, B_ROWS / 128);   // 133 x 4 = 532 blocks
    gemm_bias_lrelu_kernel<<<grid, 256, 0, stream>>>(Xb, Wb, bias, out);
}
